// BiLSTM_CRF_14980845928895
// MI455X (gfx1250) — matrix operand order repaired and checked by reading, not compiled, not run on hardware
//
#include <hip/hip_runtime.h>
#include <hip/hip_bf16.h>

#define S_  8192
#define E_  1024
#define H2_ 256
#define NG_ 1024   // 4*H2
#define T_  17

typedef __attribute__((ext_vector_type(16))) __bf16 v16bf;
typedef __attribute__((ext_vector_type(8)))  float  v8f;

union ABFrag { v16bf v; unsigned short u[16]; };

__device__ __forceinline__ unsigned short f2bf(float f) {
    unsigned u = __float_as_uint(f);
    unsigned r = u + 0x7FFFu + ((u >> 16) & 1u);   // round-to-nearest-even
    return (unsigned short)(r >> 16);
}
__device__ __forceinline__ float bf_lo(unsigned u) { return __uint_as_float(u << 16); }
__device__ __forceinline__ float bf_hi(unsigned u) { return __uint_as_float(u & 0xFFFF0000u); }
__device__ __forceinline__ float sigmoidf(float x) { return 1.0f / (1.0f + __expf(-x)); }

// ---------------------------------------------------------------- convert f32 -> bf16
__global__ void cvt_bf16_kernel(const float* __restrict__ in, unsigned short* __restrict__ out, int n) {
    int i = blockIdx.x * blockDim.x + threadIdx.x;
    int stride = gridDim.x * blockDim.x;
    for (; i < n; i += stride) out[i] = f2bf(in[i]);
}

// ---------------------------------------------------------------- bias prep
__global__ void prep_bias_kernel(const float* bihf, const float* bhhf,
                                 const float* bihb, const float* bhhb,
                                 float* biasf, float* biasb) {
    int i = blockIdx.x * blockDim.x + threadIdx.x;
    if (i < NG_) { biasf[i] = bihf[i] + bhhf[i]; biasb[i] = bihb[i] + bhhb[i]; }
}

// ---------------------------------------------------------------- WMMA NT-GEMM
// C[M,N] = A[M,K](bf16) * B[N,K](bf16)^T + bias[N], M=8192, N=K=1024.
// Block = 256 threads = 8 waves; block tile 128(M)x64(N); wave tile 16x64.
__global__ void __launch_bounds__(256)
gemm_xg_kernel(const unsigned short* __restrict__ A,
               const unsigned short* __restrict__ B,
               const float* __restrict__ bias,
               float* __restrict__ C) {
    const int K = E_, N = NG_;
    const int lane = threadIdx.x & 31;
    const int wave = threadIdx.x >> 5;
    const int m0 = blockIdx.x * 128 + wave * 16;
    const int n0 = blockIdx.y * 64;

    const int am  = lane & 15, ahi  = lane >> 4;   // A frag: row, k-half select
    const int bn  = lane & 15, bkhi = lane >> 4;   // B frag: col, k-half select

    v8f acc[4] = {v8f{}, v8f{}, v8f{}, v8f{}};

    const unsigned short* Arow = A + (size_t)(m0 + am) * K;

    for (int k0 = 0; k0 < K; k0 += 32) {
        ABFrag a;
        #pragma unroll
        for (int e = 0; e < 16; ++e) {
            // A frag k order used here: k = 16*hi + e (contiguous, matches B)
            int k = (ahi << 4) + e;
            a.u[e] = Arow[k0 + k];
        }
        ABFrag b[4];
        #pragma unroll
        for (int nt = 0; nt < 4; ++nt) {
            // B frag k order used here (same as A): lanes 0-15 K=0..15, lanes 16-31 K=16..31, contiguous in k
            const unsigned short* Brow = B + (size_t)(n0 + nt * 16 + bn) * K + k0 + (bkhi << 4);
            #pragma unroll
            for (int e = 0; e < 16; ++e) b[nt].u[e] = Brow[e];
        }
        acc[0] = __builtin_amdgcn_wmma_f32_16x16x32_bf16(false, a.v, false, b[0].v, (short)0, acc[0], false, false);
        acc[1] = __builtin_amdgcn_wmma_f32_16x16x32_bf16(false, a.v, false, b[1].v, (short)0, acc[1], false, false);
        acc[2] = __builtin_amdgcn_wmma_f32_16x16x32_bf16(false, a.v, false, b[2].v, (short)0, acc[2], false, false);
        acc[3] = __builtin_amdgcn_wmma_f32_16x16x32_bf16(false, a.v, false, b[3].v, (short)0, acc[3], false, false);
    }

    const int cn = lane & 15, cmhi = lane >> 4;
    #pragma unroll
    for (int r = 0; r < 8; ++r) {
        int m = m0 + r + (cmhi << 3);            // f32 C/D layout: VGPR r -> row r (+8 for hi lanes)
        float* crow = C + (size_t)m * N + n0 + cn;
        crow[0]  = acc[0][r] + bias[n0 + cn];
        crow[16] = acc[1][r] + bias[n0 + 16 + cn];
        crow[32] = acc[2][r] + bias[n0 + 32 + cn];
        crow[48] = acc[3][r] + bias[n0 + 48 + cn];
    }
}

// ---------------------------------------------------------------- LSTM recurrence
// One workgroup per direction (blockIdx.x: 0=fwd, 1=bwd). 1024 threads: one gate row each.
__global__ void __launch_bounds__(1024)
lstm_kernel(const float* __restrict__ xg_f, const float* __restrict__ xg_b,
            const unsigned short* __restrict__ whhf_bf, const unsigned short* __restrict__ whhb_bf,
            float* __restrict__ h_f, float* __restrict__ h_b) {
    const int dir = blockIdx.x;
    const float* xg = dir ? xg_b : xg_f;
    const unsigned short* whh = dir ? whhb_bf : whhf_bf;
    float* hout = dir ? h_b : h_f;

    __shared__ float h_s[H2_];
    __shared__ float g_s[NG_];
    const int tid = threadIdx.x;
    if (tid < H2_) h_s[tid] = 0.0f;
    float c = 0.0f;
    __syncthreads();

    const unsigned short* wrow = whh + (size_t)tid * H2_;

    for (int step = 0; step < S_; ++step) {
        const int t = dir ? (S_ - 1 - step) : step;
        float acc = xg[(size_t)t * NG_ + tid];
        #pragma unroll 4
        for (int k = 0; k < H2_; k += 8) {
            uint4 w4 = *(const uint4*)(wrow + k);
            acc += bf_lo(w4.x) * h_s[k + 0] + bf_hi(w4.x) * h_s[k + 1];
            acc += bf_lo(w4.y) * h_s[k + 2] + bf_hi(w4.y) * h_s[k + 3];
            acc += bf_lo(w4.z) * h_s[k + 4] + bf_hi(w4.z) * h_s[k + 5];
            acc += bf_lo(w4.w) * h_s[k + 6] + bf_hi(w4.w) * h_s[k + 7];
        }
        g_s[tid] = acc;
        __syncthreads();
        if (tid < H2_) {
            float ig = sigmoidf(g_s[tid]);
            float fg = sigmoidf(g_s[tid + H2_]);
            float gg = tanhf(g_s[tid + 2 * H2_]);
            float og = sigmoidf(g_s[tid + 3 * H2_]);
            c = fg * c + ig * gg;
            float h = og * tanhf(c);
            h_s[tid] = h;
            hout[(size_t)t * H2_ + tid] = h;
        }
        __syncthreads();
    }
}

// ---------------------------------------------------------------- emissions
__global__ void emissions_kernel(const float* __restrict__ h_f, const float* __restrict__ h_b,
                                 const float* __restrict__ w_out, const float* __restrict__ b_out,
                                 float* __restrict__ em) {
    const int s = blockIdx.x;
    const int t = threadIdx.x;
    if (t >= T_) return;
    const float* hf = h_f + (size_t)s * H2_;
    const float* hb = h_b + (size_t)s * H2_;
    const float* w  = w_out + (size_t)t * 512;
    float acc = b_out[t];
    #pragma unroll 4
    for (int k = 0; k < H2_; ++k) acc += hf[k] * w[k];
    #pragma unroll 4
    for (int k = 0; k < H2_; ++k) acc += hb[k] * w[H2_ + k];
    em[(size_t)s * T_ + t] = acc;
}

// ---------------------------------------------------------------- gold score (deterministic 2-stage)
__global__ void score_kernel(const float* __restrict__ em, const int* __restrict__ tags,
                             const float* __restrict__ trans, const float* __restrict__ start_t,
                             const float* __restrict__ end_t, float* __restrict__ partials) {
    const int s = blockIdx.x * blockDim.x + threadIdx.x;  // exactly covers S_
    int tg = tags[s];
    float local = em[(size_t)s * T_ + tg];
    if (s > 0) local += trans[tags[s - 1] * T_ + tg];
    else       local += start_t[tg];
    if (s == S_ - 1) local += end_t[tg];

    __shared__ float red[256];
    red[threadIdx.x] = local;
    __syncthreads();
    for (int off = 128; off > 0; off >>= 1) {
        if (threadIdx.x < off) red[threadIdx.x] += red[threadIdx.x + off];
        __syncthreads();
    }
    if (threadIdx.x == 0) partials[blockIdx.x] = red[0];
}

// ---------------------------------------------------------------- CRF forward + Viterbi + backtrack
// One wave. Backpointers packed as bytes into dynamic LDS (8192*17 = 139264 B < 320 KB).
__global__ void crf_kernel(const float* __restrict__ em, const float* __restrict__ start_t,
                           const float* __restrict__ end_t, const float* __restrict__ trans,
                           const float* __restrict__ score_partials, int n_partials,
                           int* __restrict__ path, float* __restrict__ loss_out) {
    extern __shared__ unsigned char bp_s[];
    __shared__ float alpha_s[32], v_s[32], trans_s[T_ * T_];
    const int j = threadIdx.x;

    for (int i = j; i < T_ * T_; i += 32) trans_s[i] = trans[i];
    if (j < T_) {
        float a0 = start_t[j] + em[j];
        alpha_s[j] = a0;
        v_s[j] = a0;
    }
    __syncthreads();

    for (int s = 1; s < S_; ++s) {
        float na = 0.0f, nv = 0.0f;
        int bi = 0;
        if (j < T_) {
            float e = em[(size_t)s * T_ + j];
            float m = -1e30f, bm = -1e30f;
            for (int i = 0; i < T_; ++i) {
                float tr = trans_s[i * T_ + j];
                float ta = alpha_s[i] + tr;
                m = fmaxf(m, ta);
                float tv = v_s[i] + tr;
                if (tv > bm) { bm = tv; bi = i; }
            }
            float sum = 0.0f;
            for (int i = 0; i < T_; ++i) sum += __expf(alpha_s[i] + trans_s[i * T_ + j] - m);
            na = m + __logf(sum) + e;
            nv = bm + e;
        }
        __syncthreads();
        if (j < T_) {
            alpha_s[j] = na;
            v_s[j] = nv;
            bp_s[s * T_ + j] = (unsigned char)bi;
        }
        __syncthreads();
    }

    if (j == 0) {
        float m = -1e30f;
        for (int i = 0; i < T_; ++i) m = fmaxf(m, alpha_s[i] + end_t[i]);
        float sum = 0.0f;
        for (int i = 0; i < T_; ++i) sum += __expf(alpha_s[i] + end_t[i] - m);
        float logZ = m + __logf(sum);

        float bm = -1e30f; int last = 0;
        for (int i = 0; i < T_; ++i) {
            float t = v_s[i] + end_t[i];
            if (t > bm) { bm = t; last = i; }
        }
        int tag = last;
        path[S_ - 1] = tag;
        for (int s = S_ - 1; s >= 1; --s) {
            tag = bp_s[s * T_ + tag];
            path[s - 1] = tag;
        }
        float score = 0.0f;
        for (int i = 0; i < n_partials; ++i) score += score_partials[i];
        *loss_out = -(score - logZ) * 10.0f;
    }
}

// ---------------------------------------------------------------- finalize: loss + path -> d_out (float)
__global__ void finalize_kernel(const float* __restrict__ loss, const int* __restrict__ path,
                                float* __restrict__ out) {
    int i = blockIdx.x * blockDim.x + threadIdx.x;
    if (i == 0) out[0] = loss[0];
    if (i < S_) out[1 + i] = (float)path[i];
}

// ================================================================ host
extern "C" void kernel_launch(void* const* d_in, const int* in_sizes, int n_in,
                              void* d_out, int out_size, void* d_ws, size_t ws_size,
                              hipStream_t stream) {
    (void)in_sizes; (void)n_in; (void)out_size; (void)ws_size;

    const float* x       = (const float*)d_in[0];
    const int*   tags    = (const int*)  d_in[1];
    const float* w_ih_f  = (const float*)d_in[2];
    const float* w_hh_f  = (const float*)d_in[3];
    const float* b_ih_f  = (const float*)d_in[4];
    const float* b_hh_f  = (const float*)d_in[5];
    const float* w_ih_b  = (const float*)d_in[6];
    const float* w_hh_b  = (const float*)d_in[7];
    const float* b_ih_b  = (const float*)d_in[8];
    const float* b_hh_b  = (const float*)d_in[9];
    const float* w_out   = (const float*)d_in[10];
    const float* b_out   = (const float*)d_in[11];
    const float* start_t = (const float*)d_in[12];
    const float* end_t   = (const float*)d_in[13];
    const float* trans   = (const float*)d_in[14];

    char* base = (char*)d_ws;
    size_t off = 0;
    auto alloc = [&](size_t bytes) -> void* {
        void* p = base + off;
        off += bytes;
        off = (off + 255) & ~(size_t)255;
        return p;
    };

    unsigned short* x_bf    = (unsigned short*)alloc((size_t)S_ * E_ * 2);
    unsigned short* wihf_bf = (unsigned short*)alloc((size_t)NG_ * E_ * 2);
    unsigned short* wihb_bf = (unsigned short*)alloc((size_t)NG_ * E_ * 2);
    unsigned short* whhf_bf = (unsigned short*)alloc((size_t)NG_ * H2_ * 2);
    unsigned short* whhb_bf = (unsigned short*)alloc((size_t)NG_ * H2_ * 2);
    float* bias_f = (float*)alloc(NG_ * 4);
    float* bias_b = (float*)alloc(NG_ * 4);
    float* xg_f   = (float*)alloc((size_t)S_ * NG_ * 4);
    float* xg_b   = (float*)alloc((size_t)S_ * NG_ * 4);
    float* h_f    = (float*)alloc((size_t)S_ * H2_ * 4);
    float* h_b    = (float*)alloc((size_t)S_ * H2_ * 4);
    float* em     = (float*)alloc((size_t)S_ * T_ * 4);
    float* partials = (float*)alloc(32 * 4);
    float* loss_ws  = (float*)alloc(4);
    int*   path_ws  = (int*)alloc(S_ * 4);

    // 1) conversions
    cvt_bf16_kernel<<<2048, 256, 0, stream>>>(x,      x_bf,    S_ * E_);
    cvt_bf16_kernel<<<1024, 256, 0, stream>>>(w_ih_f, wihf_bf, NG_ * E_);
    cvt_bf16_kernel<<<1024, 256, 0, stream>>>(w_ih_b, wihb_bf, NG_ * E_);
    cvt_bf16_kernel<<<256,  256, 0, stream>>>(w_hh_f, whhf_bf, NG_ * H2_);
    cvt_bf16_kernel<<<256,  256, 0, stream>>>(w_hh_b, whhb_bf, NG_ * H2_);
    prep_bias_kernel<<<4, 256, 0, stream>>>(b_ih_f, b_hh_f, b_ih_b, b_hh_b, bias_f, bias_b);

    // 2) input projections via WMMA bf16 (M=8192, N=K=1024)
    dim3 ggrid(S_ / 128, NG_ / 64);
    gemm_xg_kernel<<<ggrid, 256, 0, stream>>>(x_bf, wihf_bf, bias_f, xg_f);
    gemm_xg_kernel<<<ggrid, 256, 0, stream>>>(x_bf, wihb_bf, bias_b, xg_b);

    // 3) sequential LSTM (both directions concurrently)
    lstm_kernel<<<2, 1024, 0, stream>>>(xg_f, xg_b, whhf_bf, whhb_bf, h_f, h_b);

    // 4) emissions
    emissions_kernel<<<S_, 32, 0, stream>>>(h_f, h_b, w_out, b_out, em);

    // 5) gold-path score (deterministic two-stage)
    score_kernel<<<S_ / 256, 256, 0, stream>>>(em, tags, trans, start_t, end_t, partials);

    // 6) CRF forward + Viterbi + backtrack (backpointers in 139 KB dynamic LDS)
    crf_kernel<<<1, 32, S_ * T_, stream>>>(em, start_t, end_t, trans, partials, S_ / 256,
                                           path_ws, loss_ws);

    // 7) finalize output: [loss, path...]
    finalize_kernel<<<(S_ + 1 + 255) / 256, 256, 0, stream>>>(loss_ws, path_ws, (float*)d_out);
}
